// STAGCNBlock_50225347559491
// MI455X (gfx1250) — matrix operand order repaired and checked by reading, not compiled, not run on hardware
//
#include <hip/hip_runtime.h>
#include <hip/hip_bf16.h>

// ---------------- problem constants (match reference) ----------------
constexpr int B_    = 64;
constexpr int C_IN  = 64;
constexpr int T_    = 512;
constexpr int HEADS = 2;
constexpr int DIM_H = 64;
constexpr int KW    = 9;
constexpr int N_    = B_ * T_;          // 32768 nodes
constexpr int D_    = HEADS * DIM_H;    // 128
constexpr int E_MAX = 524288;           // reference edge count

typedef __attribute__((ext_vector_type(16))) _Float16 v16h;
typedef __attribute__((ext_vector_type(8)))  float    v8f;

#define CDIV(a, b) (((a) + (b) - 1) / (b))

// ---------------- workspace layout (bytes) ----------------
constexpr size_t OFF_SCALE0 = 0;
constexpr size_t OFF_SHIFT0 = 256;
constexpr size_t OFF_SCALE1 = 512;
constexpr size_t OFF_SHIFT1 = 1024;
constexpr size_t OFF_SCALE2 = 1536;
constexpr size_t OFF_SHIFT2 = 3584;
constexpr size_t OFF_XNH  = 8192;                                       // N*64 f16   = 4 MiB
constexpr size_t OFF_WLH  = OFF_XNH  + (size_t)N_ * 64 * 2;             // frag-order, 16 KiB
constexpr size_t OFF_WRH  = OFF_WLH  + 64 * 128 * 2;                    // frag-order, 16 KiB
constexpr size_t OFF_WCH  = OFF_WRH  + 64 * 128 * 2;                    // frag-order, 4.5 MiB
constexpr size_t OFF_XL   = OFF_WCH  + (size_t)512 * 512 * 9 * 2;       // N*128 f32  = 16 MiB
constexpr size_t OFF_XR   = OFF_XL   + (size_t)N_ * 128 * 4;            // 16 MiB (reused as F)
constexpr size_t OFF_GOUT = OFF_XR   + (size_t)N_ * 128 * 4;            // 16 MiB (reused as conv_raw)
constexpr size_t OFF_EBUF = OFF_GOUT + (size_t)N_ * 128 * 4;            // (E+N)*2 f32
constexpr size_t OFF_MAXB = OFF_EBUF + (size_t)(E_MAX + N_) * 2 * 4;    // N*2 u32
constexpr size_t OFF_ZB   = OFF_MAXB + (size_t)N_ * 2 * 4;              // N*2 f32
constexpr size_t OFF_XTB  = OFF_ZB   + (size_t)N_ * 2 * 4;              // B*512*128 f16 = 8 MiB

// ---------------- helpers ----------------
__device__ __forceinline__ unsigned f2ord(float f) {
  unsigned u = __float_as_uint(f);
  return (u & 0x80000000u) ? ~u : (u | 0x80000000u);
}
__device__ __forceinline__ float ord2f(unsigned u) {
  return (u & 0x80000000u) ? __uint_as_float(u & 0x7fffffffu) : __uint_as_float(~u);
}

// WMMA wrapper (codegen-confirmed builtin): D = A(16x32 f16) * B(32x16 f16) + C(f32)
__device__ __forceinline__ v8f wmma32(v16h a, v16h b, v8f c) {
  return __builtin_amdgcn_wmma_f32_16x16x32_f16(false, a, false, b, (short)0, c, false, false);
}
// A 16x32 layout: element i of the per-lane v16h holds K = i (+16 for high half-lanes):
__device__ __forceinline__ int a_kidx(int i, int hi) {
  return i + (hi ? 16 : 0);
}
// B 32x16: element i holds K = i (+16 for high half-lanes), N = lane&15
__device__ __forceinline__ int b_kidx(int i, int hi) { return i + (hi ? 16 : 0); }

// ---------------- BN statistics (fused scale/shift) ----------------
__global__ __launch_bounds__(256) void bn0_stats(const float* __restrict__ x,
    const float* __restrict__ g, const float* __restrict__ bb,
    float* __restrict__ scale, float* __restrict__ shift) {
  __shared__ float ss[256], sq[256];
  int c = blockIdx.x, tid = threadIdx.x;
  float s = 0.f, q = 0.f;
  for (int j = tid; j < B_ * T_; j += 256) {
    int bi = j >> 9, t = j & 511;
    float v = x[bi * (C_IN * T_) + c * T_ + t];
    s += v; q += v * v;
  }
  ss[tid] = s; sq[tid] = q; __syncthreads();
  for (int st = 128; st; st >>= 1) {
    if (tid < st) { ss[tid] += ss[tid + st]; sq[tid] += sq[tid + st]; }
    __syncthreads();
  }
  if (tid == 0) {
    float m = ss[0] / (float)(B_ * T_);
    float v = sq[0] / (float)(B_ * T_) - m * m;
    float sc = g[c] * rsqrtf(v + 1e-5f);
    scale[c] = sc; shift[c] = bb[c] - m * sc;
  }
}

__global__ __launch_bounds__(256) void bn1_stats(const float* __restrict__ y,
    const float* __restrict__ g, const float* __restrict__ bb,
    float* __restrict__ scale, float* __restrict__ shift) {
  __shared__ float ss[256], sq[256];
  int d = blockIdx.x, tid = threadIdx.x;
  float s = 0.f, q = 0.f;
  for (int j = tid; j < N_; j += 256) { float v = y[j * D_ + d]; s += v; q += v * v; }
  ss[tid] = s; sq[tid] = q; __syncthreads();
  for (int st = 128; st; st >>= 1) {
    if (tid < st) { ss[tid] += ss[tid + st]; sq[tid] += sq[tid + st]; }
    __syncthreads();
  }
  if (tid == 0) {
    float m = ss[0] / (float)N_;
    float v = sq[0] / (float)N_ - m * m;
    float sc = g[d] * rsqrtf(v + 1e-5f);
    scale[d] = sc; shift[d] = bb[d] - m * sc;
  }
}

__global__ __launch_bounds__(256) void bn2_stats(const float* __restrict__ y,
    const float* __restrict__ g, const float* __restrict__ bb,
    float* __restrict__ scale, float* __restrict__ shift) {
  __shared__ float ss[256], sq[256];
  int o = blockIdx.x, tid = threadIdx.x;
  float s = 0.f, q = 0.f;
  for (int j = tid; j < B_ * D_; j += 256) {
    int bi = j >> 7, l = j & 127;
    float v = y[bi * (T_ * D_) + o * D_ + l];
    s += v; q += v * v;
  }
  ss[tid] = s; sq[tid] = q; __syncthreads();
  for (int st = 128; st; st >>= 1) {
    if (tid < st) { ss[tid] += ss[tid + st]; sq[tid] += sq[tid + st]; }
    __syncthreads();
  }
  if (tid == 0) {
    float m = ss[0] / (float)(B_ * D_);
    float v = sq[0] / (float)(B_ * D_) - m * m;
    float sc = g[o] * rsqrtf(v + 1e-5f);
    scale[o] = sc; shift[o] = bb[o] - m * sc;
  }
}

// ---------------- elementwise / packing ----------------
__global__ __launch_bounds__(256) void bn0_apply_pack(const float* __restrict__ x,
    const float* __restrict__ scale, const float* __restrict__ shift,
    _Float16* __restrict__ xnh) {
  int i = blockIdx.x * 256 + threadIdx.x;
  if (i >= B_ * C_IN * T_) return;
  int bi = i / (C_IN * T_); int r = i % (C_IN * T_);
  int c = r >> 9; int t = r & 511;
  float v = x[i] * scale[c] + shift[c];
  xnh[(size_t)(bi * T_ + t) * C_IN + c] = (_Float16)v;   // [N, C] for GEMM A
}

// W_l / W_r converted straight into per-lane WMMA B-fragment order:
//   idx = ((ntile*2 + chunk)*32 + lane)*16 + elem
//   value = W[chunk*32 + b_kidx(elem, hi)][ntile*16 + l16]
__global__ __launch_bounds__(256) void cvt_w(const float* __restrict__ wl,
    const float* __restrict__ wr, _Float16* __restrict__ wlh, _Float16* __restrict__ wrh) {
  int i = blockIdx.x * 256 + threadIdx.x;
  if (i >= 2 * C_IN * D_) return;
  int mat = i / (C_IN * D_);
  int r = i % (C_IN * D_);
  int elem = r & 15, lane = (r >> 4) & 31, chunk = (r >> 9) & 1, ntile = r >> 10;
  int hi = lane >> 4, l16 = lane & 15;
  int row = chunk * 32 + b_kidx(elem, hi);
  float v = (mat == 0) ? wl[row * D_ + ntile * 16 + l16] : wr[row * D_ + ntile * 16 + l16];
  ((mat == 0) ? wlh : wrh)[r] = (_Float16)v;
}

// conv weights into per-lane WMMA A-fragment order:
//   idx = (((k*32 + otile)*16 + ibc)*32 + lane)*16 + elem
//   value = conv_w[o = otile*16 + l16][ic = ibc*32 + a_kidx(elem, hi)][k]
__global__ __launch_bounds__(256) void cvt_convw(const float* __restrict__ cw,
    _Float16* __restrict__ wch) {
  int i = blockIdx.x * 256 + threadIdx.x;
  if (i >= T_ * T_ * KW) return;
  int elem  = i & 15;
  int lane  = (i >> 4) & 31;
  int ibc   = (i >> 9) & 15;
  int otile = (i >> 13) & 31;
  int k     = i >> 18;
  int hi = lane >> 4, l16 = lane & 15;
  int o  = otile * 16 + l16;
  int ic = ibc * 32 + a_kidx(elem, hi);
  wch[i] = (_Float16)cw[((size_t)o * 512 + ic) * KW + k];
}

__global__ __launch_bounds__(256) void init_gat(float* __restrict__ gout,
    const float* __restrict__ gat_b, unsigned* __restrict__ maxb, float* __restrict__ zb) {
  int i = blockIdx.x * 256 + threadIdx.x;
  if (i < N_ * HEADS) { maxb[i] = 0x007FFFFFu /* ord(-inf) */; zb[i] = 0.f; }
  if (i < N_ * D_) gout[i] = gat_b[i & (D_ - 1)];
}

// ---------------- GATv2 linear transforms: xl = xn*W_l, xr = xn*W_r ----------------
// 512 threads = 16 waves: waves 0..7 -> W_l ntiles 0..7, waves 8..15 -> W_r.
// A-tile staged in LDS already in fragment order (A frag depends only on lane),
// B fragments are contiguous v16h in the pre-swizzled weight buffers.
__global__ __launch_bounds__(512) void gemm_xlr(const _Float16* __restrict__ xnh,
    const _Float16* __restrict__ wlh, const _Float16* __restrict__ wrh,
    float* __restrict__ xl, float* __restrict__ xr) {
  __shared__ _Float16 sA[2 * 32 * 16];          // [chunk][lane][elem], 2 KiB
  int tid = threadIdx.x;
  int mtile = blockIdx.x;
  for (int i = tid; i < 1024; i += 512) {
    int elem = i & 15, lane = (i >> 4) & 31, chunk = i >> 9;
    int hi = lane >> 4, l16 = lane & 15;
    sA[i] = xnh[(size_t)(mtile * 16 + l16) * C_IN + chunk * 32 + a_kidx(elem, hi)];
  }
  __syncthreads();
  int wave = tid >> 5, lane = tid & 31;
  int hi = (lane >> 4) & 1, l16 = lane & 15;
  int ntile = wave & 7;
  const _Float16* W = (wave < 8) ? wlh : wrh;
  float* out = (wave < 8) ? xl : xr;
  v8f c = {};
#pragma unroll
  for (int chunk = 0; chunk < 2; ++chunk) {
    v16h a = *(const v16h*)&sA[(chunk * 32 + lane) * 16];
    v16h b = *(const v16h*)&W[((ntile * 2 + chunk) * 32 + lane) * 16];
    c = wmma32(a, b, c);
  }
  for (int v = 0; v < 8; ++v) {
    int m = v + (hi ? 8 : 0);
    out[(size_t)(mtile * 16 + m) * D_ + ntile * 16 + l16] = c[v];
  }
}

// ---------------- GATv2 edge phase ----------------
__global__ __launch_bounds__(256) void edge_logits(const int* __restrict__ ei,
    const float* __restrict__ xl, const float* __restrict__ xr,
    const float* __restrict__ att, float* __restrict__ ebuf,
    unsigned* __restrict__ maxb, int E, int EN) {
  int idx = blockIdx.x * 256 + threadIdx.x;
  if (idx >= EN * 2) return;
  int e = idx >> 1, h = idx & 1;
  int src = (e < E) ? ei[e] : (e - E);
  int dst = (e < E) ? ei[E + e] : (e - E);
  const float* pl = xl + (size_t)src * D_ + h * DIM_H;
  const float* pr = xr + (size_t)dst * D_ + h * DIM_H;
  const float* pa = att + h * DIM_H;
  float acc = 0.f;
  for (int d = 0; d < DIM_H; ++d) {
    float v = pl[d] + pr[d];
    v = (v > 0.f) ? v : 0.2f * v;       // leaky_relu(0.2)
    acc = fmaf(v, pa[d], acc);
  }
  ebuf[idx] = acc;
  atomicMax(&maxb[dst * 2 + h], f2ord(acc));
}

__global__ __launch_bounds__(256) void edge_exp(const int* __restrict__ ei,
    float* __restrict__ ebuf, const unsigned* __restrict__ maxb,
    float* __restrict__ zb, int E, int EN) {
  int idx = blockIdx.x * 256 + threadIdx.x;
  if (idx >= EN * 2) return;
  int e = idx >> 1, h = idx & 1;
  int dst = (e < E) ? ei[E + e] : (e - E);
  float a = __expf(ebuf[idx] - ord2f(maxb[dst * 2 + h]));
  ebuf[idx] = a;
  atomicAdd(&zb[dst * 2 + h], a);
}

__global__ __launch_bounds__(256) void edge_agg(const int* __restrict__ ei,
    const float* __restrict__ ebuf, const float* __restrict__ zb,
    const float* __restrict__ xl, float* __restrict__ gout, int E, int EN) {
  long long idx = (long long)blockIdx.x * 256 + threadIdx.x;
  if (idx >= (long long)EN * 32) return;
  int e = (int)(idx >> 5); int s = (int)(idx & 31);
  int h = s >> 4; int d0 = (s & 15) * 4;
  int src = (e < E) ? ei[e] : (e - E);
  int dst = (e < E) ? ei[E + e] : (e - E);
  float alpha = ebuf[e * 2 + h] / zb[dst * 2 + h];
  const float* pl = xl + (size_t)src * D_ + h * DIM_H + d0;
  float* po = gout + (size_t)dst * D_ + h * DIM_H + d0;
  for (int j = 0; j < 4; ++j) atomicAdd(&po[j], alpha * pl[j]);
}

// ---------------- BN1 + ReLU -> residual F (f32) and conv input xtb (f16) ----------------
__global__ __launch_bounds__(256) void bn1_apply(const float* __restrict__ gout,
    const float* __restrict__ scale, const float* __restrict__ shift,
    float* __restrict__ F, _Float16* __restrict__ xtb) {
  int i = blockIdx.x * 256 + threadIdx.x;
  if (i >= N_ * D_) return;
  int d = i & (D_ - 1);
  float v = gout[i] * scale[d] + shift[d];
  v = (v > 0.f) ? v : 0.f;
  F[i] = v;                                     // flat == x3 (raw reshape, residual)
  int bi = i / (D_ * T_); int r = i % (D_ * T_);
  int m = r >> 9; int ic = r & 511;             // x3[b, m, ic] -> xt[b, ic, m]
  xtb[(size_t)bi * (T_ * D_) + ic * D_ + m] = (_Float16)v;
}

// ---------------- conv as implicit GEMM (dominant compute) ----------------
// grid (4, 8, 64), 256 threads = 8 waves; wave -> otile, blockIdx.y -> ltile, blockIdx.z -> b.
// Activation slab staged TRANSPOSED in LDS: sXT[mm*512 + ic] so each B fragment is a
// contiguous v16h (2x ds_load_b128). Weights pre-swizzled -> contiguous v16h global loads.
// Inner loop: 1 global v16h + 1 LDS v16h + 1 WMMA; 144 WMMAs per wave.
__global__ __launch_bounds__(256) void conv_gemm(const _Float16* __restrict__ wch,
    const _Float16* __restrict__ xtb, const float* __restrict__ conv_b,
    float* __restrict__ y) {
  __shared__ _Float16 sXT[24 * 512];            // [mm][ic], 24 KiB (16 cols + 8 halo)
  int tid = threadIdx.x;
  int b = blockIdx.z, ltile = blockIdx.y;
  int mbase = ltile * 16 - 4;                   // SAME pad = 4
  const _Float16* xb = xtb + (size_t)b * (T_ * D_);
  for (int i = tid; i < 512 * 24; i += 256) {
    int ic = i / 24, mm = i % 24;               // coalesced-ish global reads, scattered LDS writes
    int m = mbase + mm;
    sXT[mm * 512 + ic] = (m >= 0 && m < D_) ? xb[ic * D_ + m] : (_Float16)0.0f;
  }
  __syncthreads();
  int wave = tid >> 5, lane = tid & 31;
  int hi = (lane >> 4) & 1, l16 = lane & 15;
  int otile = blockIdx.x * 8 + wave;
  v8f c = {};
  for (int k = 0; k < KW; ++k) {
    const _Float16* wb = wch + ((size_t)(k * 32 + otile) * 512 + lane) * 16;
    const _Float16* srow = &sXT[(l16 + k) * 512 + hi * 16];
    if (k + 1 < KW)
      __builtin_prefetch(wch + ((size_t)((k + 1) * 32 + otile) * 512 + lane) * 16, 0, 1);
#pragma unroll
    for (int ibc = 0; ibc < 16; ++ibc) {
      v16h a = *(const v16h*)(wb + ibc * 512);      // 32B contiguous fragment
      v16h bf = *(const v16h*)(srow + ibc * 32);    // 32B contiguous from LDS
      c = wmma32(a, bf, c);
    }
  }
  for (int v = 0; v < 8; ++v) {
    int o = otile * 16 + v + (hi ? 8 : 0);
    int l = ltile * 16 + l16;
    y[(size_t)b * (T_ * D_) + o * D_ + l] = c[v] + conv_b[o];
  }
}

// ---------------- final: residual + relu(bn2(conv)) ----------------
__global__ __launch_bounds__(256) void final_add(const float* __restrict__ F,
    const float* __restrict__ craw, const float* __restrict__ scale,
    const float* __restrict__ shift, float* __restrict__ out) {
  int i = blockIdx.x * 256 + threadIdx.x;
  if (i >= N_ * D_) return;
  int bi = i / (D_ * T_); int r = i % (D_ * T_);
  int d = r >> 9; int t = r & 511;              // out[b, d, t] += ybn[b, t, d]
  float y = craw[(size_t)bi * (T_ * D_) + t * D_ + d] * scale[t] + shift[t];
  y = (y > 0.f) ? y : 0.f;
  out[i] = F[i] + y;
}

// ---------------- host launcher ----------------
extern "C" void kernel_launch(void* const* d_in, const int* in_sizes, int n_in,
                              void* d_out, int out_size, void* d_ws, size_t ws_size,
                              hipStream_t stream) {
  const float* x      = (const float*)d_in[0];
  const int*   ei     = (const int*)  d_in[1];
  const float* bn0_g  = (const float*)d_in[2];
  const float* bn0_b  = (const float*)d_in[3];
  const float* W_l    = (const float*)d_in[4];
  const float* W_r    = (const float*)d_in[5];
  const float* att    = (const float*)d_in[6];
  const float* gat_b  = (const float*)d_in[7];
  const float* bn1_g  = (const float*)d_in[8];
  const float* bn1_b  = (const float*)d_in[9];
  const float* conv_w = (const float*)d_in[10];
  const float* conv_b = (const float*)d_in[11];
  const float* bn2_g  = (const float*)d_in[12];
  const float* bn2_b  = (const float*)d_in[13];
  int E = in_sizes[1] / 2;

  char* w = (char*)d_ws;
  float*     scale0 = (float*)(w + OFF_SCALE0);
  float*     shift0 = (float*)(w + OFF_SHIFT0);
  float*     scale1 = (float*)(w + OFF_SCALE1);
  float*     shift1 = (float*)(w + OFF_SHIFT1);
  float*     scale2 = (float*)(w + OFF_SCALE2);
  float*     shift2 = (float*)(w + OFF_SHIFT2);
  _Float16*  xnh    = (_Float16*)(w + OFF_XNH);
  _Float16*  wlh    = (_Float16*)(w + OFF_WLH);
  _Float16*  wrh    = (_Float16*)(w + OFF_WRH);
  _Float16*  wch    = (_Float16*)(w + OFF_WCH);
  float*     xl     = (float*)(w + OFF_XL);
  float*     xr     = (float*)(w + OFF_XR);
  float*     Fbuf   = (float*)(w + OFF_XR);     // alias: xr dead after edge_logits
  float*     gout   = (float*)(w + OFF_GOUT);
  float*     craw   = (float*)(w + OFF_GOUT);   // alias: gout dead after bn1_apply
  float*     ebuf   = (float*)(w + OFF_EBUF);
  unsigned*  maxb   = (unsigned*)(w + OFF_MAXB);
  float*     zb     = (float*)(w + OFF_ZB);
  _Float16*  xtb    = (_Float16*)(w + OFF_XTB);

  int EN = E + N_;

  bn0_stats<<<C_IN, 256, 0, stream>>>(x, bn0_g, bn0_b, scale0, shift0);
  bn0_apply_pack<<<CDIV(B_ * C_IN * T_, 256), 256, 0, stream>>>(x, scale0, shift0, xnh);
  cvt_w<<<CDIV(2 * C_IN * D_, 256), 256, 0, stream>>>(W_l, W_r, wlh, wrh);
  cvt_convw<<<CDIV(T_ * T_ * KW, 256), 256, 0, stream>>>(conv_w, wch);
  init_gat<<<CDIV(N_ * D_, 256), 256, 0, stream>>>(gout, gat_b, maxb, zb);

  gemm_xlr<<<N_ / 16, 512, 0, stream>>>(xnh, wlh, wrh, xl, xr);

  edge_logits<<<CDIV(EN * 2, 256), 256, 0, stream>>>(ei, xl, xr, att, ebuf, maxb, E, EN);
  edge_exp<<<CDIV(EN * 2, 256), 256, 0, stream>>>(ei, ebuf, maxb, zb, E, EN);
  edge_agg<<<(int)CDIV((long long)EN * 32, 256), 256, 0, stream>>>(ei, ebuf, zb, xl, gout, E, EN);

  bn1_stats<<<D_, 256, 0, stream>>>(gout, bn1_g, bn1_b, scale1, shift1);
  bn1_apply<<<CDIV(N_ * D_, 256), 256, 0, stream>>>(gout, scale1, shift1, Fbuf, xtb);

  conv_gemm<<<dim3(4, 8, 64), 256, 0, stream>>>(wch, xtb, conv_b, craw);

  bn2_stats<<<T_, 256, 0, stream>>>(craw, bn2_g, bn2_b, scale2, shift2);
  final_add<<<CDIV(N_ * D_, 256), 256, 0, stream>>>(Fbuf, craw, scale2, shift2, (float*)d_out);
}
